// DeformableTransformer_9371618640469
// MI455X (gfx1250) — matrix operand order repaired and checked by reading, not compiled, not run on hardware
//
#include <hip/hip_runtime.h>
#include <hip/hip_bf16.h>

#define D_ 256
#define HEADS_ 8
#define HD_ 32
#define LVLS_ 4
#define PTS_ 4
#define DFFN_ 1024
#define NQ_ 300
#define B_ 2
#define LTOT_ 8500   // 80*80+40*40+20*20+10*10

__constant__ int c_lvlH[4] = {80, 40, 20, 10};
__constant__ int c_lvlW[4] = {80, 40, 20, 10};
__constant__ int c_lvlS[4] = {0, 6400, 8000, 8400};

typedef __attribute__((ext_vector_type(16))) __bf16 bf16x16;
typedef __attribute__((ext_vector_type(8)))  float  floatx8;
typedef __attribute__((ext_vector_type(4)))  unsigned int uint32x4;
typedef __attribute__((ext_vector_type(8)))  int  int32x8;
typedef __attribute__((ext_vector_type(4)))  int  int32x4;

// ---- TDM availability / arity hedging -------------------------------------
#if defined(__has_builtin)
# if __has_builtin(__builtin_amdgcn_tensor_load_to_lds)
#  define HAVE_TDM 1
# else
#  define HAVE_TDM 0
# endif
#else
# define HAVE_TDM 0
#endif
#if defined(__has_include)
# if __has_include(<hip/amd_detail/amd_gfx1250_TDM.h>)
#  define TDM_ARITY6 1
# else
#  define TDM_ARITY6 0
# endif
#else
# define TDM_ARITY6 0
#endif

#if HAVE_TDM
// 2D bf16 tile load via Tensor Data Mover.
// Descriptor per CDNA5 ISA:
//  group0: count=1 | lds_addr | global_addr(57b) | type=2
//  group1: data_size=2B, pad_enable, pad_interval=3 (16 DWORDs = 64B rows),
//          pad_amount=3 (4 DWORDs = 16B pad), tensor/tile dims, dim0 stride.
__device__ inline void tdm_load_tile(unsigned lds_addr, const __bf16* gptr,
                                     unsigned tensor_d0, unsigned tensor_d1,
                                     unsigned tile_d0, unsigned tile_d1,
                                     unsigned stride0)
{
    unsigned long long ga = (unsigned long long)(size_t)gptr;
    uint32x4 g0;
    g0.x = 1u;                                        // count = 1 valid descriptor
    g0.y = lds_addr;                                  // LDS byte address
    g0.z = (unsigned)(ga & 0xffffffffull);            // global_addr[31:0]
    g0.w = (unsigned)((ga >> 32) & 0x1ffffffull) | 0x80000000u; // [56:32] | type=2
    int32x8 g1;
    g1[0] = (int)((1u << 16) | (1u << 20) | (3u << 22) | (3u << 25));
    g1[1] = (int)((tensor_d0 & 0xffffu) << 16);
    g1[2] = (int)((tensor_d0 >> 16) | ((tensor_d1 & 0xffffu) << 16));
    g1[3] = (int)((tensor_d1 >> 16) | (tile_d0 << 16));
    g1[4] = (int)tile_d1;                             // tile_dim1 (tile_dim2=0)
    g1[5] = (int)stride0;                             // tensor_dim0_stride[31:0]
    g1[6] = 0;
    g1[7] = 0;
    int32x4 z4 = {0, 0, 0, 0};
#if TDM_ARITY6
    int32x8 z8 = {0, 0, 0, 0, 0, 0, 0, 0};
    __builtin_amdgcn_tensor_load_to_lds(g0, g1, z4, z4, z8, 0);
#else
    __builtin_amdgcn_tensor_load_to_lds(g0, g1, z4, z4, 0);
#endif
}
#endif // HAVE_TDM

// ---------------------------------------------------------------------------
// WMMA GEMM on pre-packed bf16 operands:
//   C[M,N] = act(A16[M,K] @ W16^T + bias), W16 stored [N,K] row-major.
// Tile 128x64, BK=32. 256 threads = 8 waves (4 along M x 2 along N),
// each wave -> 32x32 via 2x2 fragments of v_wmma_f32_16x16x32_bf16.
// Tiles double-buffered in LDS; staged by the Tensor Data Mover with
// s_wait_tensorcnt 2 pipelining (TDM completes in issue order), so tile t+1
// is in flight while tile t feeds the WMMAs. Cooperative-copy fallback.
// ---------------------------------------------------------------------------
#define BM 128
#define BN 64
#define BK 32
#define LDP 40   // LDS row stride in halves: 80B = 5x16B (aligned, conflict-free)

__global__ __launch_bounds__(256)
void gemm_wmma(const __bf16* __restrict__ A, const __bf16* __restrict__ Wt,
               const float* __restrict__ bias, float* __restrict__ C,
               int M, int N, int K, int act)
{
    __shared__ __bf16 As[2][BM][LDP];
    __shared__ __bf16 Bs[2][BN][LDP];

    const int tid   = threadIdx.x;
    const int lane  = tid & 31;
    const int wid   = tid >> 5;
    const int waveM = (wid & 3) * 32;   // 0,32,64,96
    const int waveN = (wid >> 2) * 32;  // 0,32
    const int m0 = blockIdx.y * BM;
    const int n0 = blockIdx.x * BN;

    floatx8 acc[2][2];
    for (int mi = 0; mi < 2; ++mi)
        for (int ni = 0; ni < 2; ++ni)
            acc[mi][ni] = (floatx8)0.0f;

    const int r  = lane & 15;
    const int hi = lane >> 4;
    const int T  = K / BK;

#if HAVE_TDM
    unsigned ldsA[2], ldsB[2];
    ldsA[0] = (unsigned)(size_t)(void*)&As[0][0][0];
    ldsA[1] = (unsigned)(size_t)(void*)&As[1][0][0];
    ldsB[0] = (unsigned)(size_t)(void*)&Bs[0][0][0];
    ldsB[1] = (unsigned)(size_t)(void*)&Bs[1][0][0];
    if (wid == 0) {
        // prologue: tile pair 0 -> buffer 0 (tensor_dim1=M zero-fills M edge)
        tdm_load_tile(ldsA[0], A  + (size_t)m0 * K, (unsigned)K, (unsigned)M, BK, BM, (unsigned)K);
        tdm_load_tile(ldsB[0], Wt + (size_t)n0 * K, (unsigned)K, (unsigned)N, BK, BN, (unsigned)K);
    }
#endif

    for (int t = 0; t < T; ++t) {
        const int cur = t & 1;
        const int k0  = t * BK;
#if HAVE_TDM
        if (wid == 0) {
            if (t + 1 < T) {
                // issue tile pair t+1 into the other buffer (its last readers
                // finished at the end-of-iteration barrier of t-1)
                tdm_load_tile(ldsA[cur ^ 1], A  + (size_t)m0 * K + k0 + BK,
                              (unsigned)K, (unsigned)M, BK, BM, (unsigned)K);
                tdm_load_tile(ldsB[cur ^ 1], Wt + (size_t)n0 * K + k0 + BK,
                              (unsigned)K, (unsigned)N, BK, BN, (unsigned)K);
                // in-order TDM: <=2 outstanding means pair t has landed
                __builtin_amdgcn_s_wait_tensorcnt(2);
            } else {
                __builtin_amdgcn_s_wait_tensorcnt(0);
            }
        }
#else
        {
            // A tile: 128 rows x 32 halves, 16B chunks, 2 per thread.
            for (int p = 0; p < 2; ++p) {
                int id  = tid + p * 256;
                int row = id >> 2;
                int c8  = (id & 3) * 8;
                uint4 v = make_uint4(0u, 0u, 0u, 0u);
                if (m0 + row < M)
                    v = *(const uint4*)(A + (size_t)(m0 + row) * K + k0 + c8);
                *(uint4*)&As[cur][row][c8] = v;
            }
            // B tile: 64 rows x 32 halves, 1 chunk per thread.
            {
                int row = tid >> 2;
                int c8  = (tid & 3) * 8;
                *(uint4*)&Bs[cur][row][c8] =
                    *(const uint4*)(Wt + (size_t)(n0 + row) * K + k0 + c8);
            }
        }
#endif
        __syncthreads();   // tile t visible to all waves

        // ---- load fragments per wave32 VGPR layouts ----
        bf16x16 afrag[2], bfrag[2];
        for (int mi = 0; mi < 2; ++mi) {
            const __bf16* ap = &As[cur][waveM + mi * 16 + r][hi * 8];
            // lanes 0-15: K{0..7,16..23}; lanes 16-31: K{8..15,24..31}
            for (int e = 0; e < 8; ++e) {
                afrag[mi][e]     = ap[e];
                afrag[mi][e + 8] = ap[e + 16];
            }
        }
        for (int ni = 0; ni < 2; ++ni) {
            const __bf16* bp = &Bs[cur][waveN + ni * 16 + r][hi * 8];
            for (int e = 0; e < 8; ++e) {
                bfrag[ni][e]     = bp[e];
                bfrag[ni][e + 8] = bp[e + 16];
            }
        }
        for (int mi = 0; mi < 2; ++mi)
            for (int ni = 0; ni < 2; ++ni)
                acc[mi][ni] = __builtin_amdgcn_wmma_f32_16x16x32_bf16(
                    false, afrag[mi], false, bfrag[ni],
                    (short)0, acc[mi][ni], false, false);

        __syncthreads();   // buffer cur free for reuse at t+2
    }

    // ---- epilogue: bias + activation, guarded store ----
    for (int mi = 0; mi < 2; ++mi) {
        for (int ni = 0; ni < 2; ++ni) {
            int col = n0 + waveN + ni * 16 + r;
            float bv = bias ? bias[col] : 0.f;
            for (int e = 0; e < 8; ++e) {
                int row = m0 + waveM + mi * 16 + e + hi * 8;
                if (row < M) {
                    float v = acc[mi][ni][e] + bv;
                    if (act == 1) v = fmaxf(v, 0.f);
                    C[(size_t)row * N + col] = v;
                }
            }
        }
    }
}

// ---------------------------------------------------------------------------
// bf16 packing kernels
// ---------------------------------------------------------------------------
__global__ void k_cvt_a(const float* __restrict__ a, __bf16* __restrict__ o, size_t n)
{
    size_t i = (size_t)blockIdx.x * blockDim.x + threadIdx.x;
    size_t stride = (size_t)gridDim.x * blockDim.x;
    for (; i < n; i += stride) o[i] = (__bf16)a[i];
}

// Pack weight to bf16 in [N][K] layout. wt==0: src is [K][N]; wt==1: src is [N][K].
__global__ void k_cvt_w(const float* __restrict__ w, __bf16* __restrict__ o,
                        int K, int N, int wt)
{
    size_t i = (size_t)blockIdx.x * blockDim.x + threadIdx.x;
    size_t total = (size_t)K * N;
    size_t stride = (size_t)gridDim.x * blockDim.x;
    for (; i < total; i += stride) {
        int k = (int)(i % K);
        int n = (int)(i / K);
        float v = wt ? w[(size_t)n * K + k] : w[(size_t)k * N + n];
        o[i] = (__bf16)v;
    }
}

// ---------------------------------------------------------------------------
// Elementwise helpers
// ---------------------------------------------------------------------------
__global__ void k_add(float* __restrict__ o, const float* __restrict__ a,
                      const float* __restrict__ b, size_t n)
{
    size_t i = (size_t)blockIdx.x * blockDim.x + threadIdx.x;
    size_t stride = (size_t)gridDim.x * blockDim.x;
    for (; i < n; i += stride) o[i] = a[i] + b[i];
}

__global__ void k_copy(float* __restrict__ o, const float* __restrict__ a, size_t n)
{
    size_t i = (size_t)blockIdx.x * blockDim.x + threadIdx.x;
    size_t stride = (size_t)gridDim.x * blockDim.x;
    for (; i < n; i += stride) o[i] = a[i];
}

__global__ void k_broadcast(float* __restrict__ o, const float* __restrict__ src,
                            size_t per, size_t n)
{
    size_t i = (size_t)blockIdx.x * blockDim.x + threadIdx.x;
    size_t stride = (size_t)gridDim.x * blockDim.x;
    for (; i < n; i += stride) o[i] = src[i % per];
}

__device__ inline float wave_sum(float v)
{
    for (int off = 16; off > 0; off >>= 1) v += __shfl_xor(v, off, 32);
    return v;
}

// y = LN(x + t) * g + b, written back into x. One wave (32 lanes) per row of 256.
__global__ __launch_bounds__(256)
void k_ln_residual(float* __restrict__ x, const float* __restrict__ t,
                   const float* __restrict__ g, const float* __restrict__ b, int M)
{
    int row  = blockIdx.x * 8 + (threadIdx.x >> 5);
    int lane = threadIdx.x & 31;
    if (row >= M) return;
    float vals[8];
    float s = 0.f;
    for (int i = 0; i < 8; ++i) {
        int c = i * 32 + lane;
        float v = x[(size_t)row * D_ + c] + t[(size_t)row * D_ + c];
        vals[i] = v; s += v;
    }
    s = wave_sum(s);
    float mu = s * (1.0f / D_);
    float s2 = 0.f;
    for (int i = 0; i < 8; ++i) { float d = vals[i] - mu; s2 += d * d; }
    s2 = wave_sum(s2);
    float rstd = rsqrtf(s2 * (1.0f / D_) + 1e-5f);
    for (int i = 0; i < 8; ++i) {
        int c = i * 32 + lane;
        x[(size_t)row * D_ + c] = (vals[i] - mu) * rstd * g[c] + b[c];
    }
}

// softmax over contiguous groups of 16 (LVLS*PTS) — one thread per group
__global__ void k_softmax16(float* __restrict__ aw, int n)
{
    int t = blockIdx.x * blockDim.x + threadIdx.x;
    if (t >= n) return;
    float* p = aw + (size_t)t * 16;
    float mx = p[0];
    for (int i = 1; i < 16; ++i) mx = fmaxf(mx, p[i]);
    float sm = 0.f;
    float e[16];
    for (int i = 0; i < 16; ++i) { e[i] = __expf(p[i] - mx); sm += e[i]; }
    float inv = 1.f / sm;
    for (int i = 0; i < 16; ++i) p[i] = e[i] * inv;
}

// ---------------------------------------------------------------------------
// Flatten src levels -> x[B,L,D]; pos + level_embed -> pos[B,L,D]
// ---------------------------------------------------------------------------
__global__ void k_flatten(const float* s0, const float* s1, const float* s2, const float* s3,
                          const float* p0, const float* p1, const float* p2, const float* p3,
                          const float* lvl_emb, float* __restrict__ x, float* __restrict__ pos)
{
    size_t t = (size_t)blockIdx.x * blockDim.x + threadIdx.x;
    size_t total = (size_t)B_ * LTOT_ * D_;
    if (t >= total) return;
    int c = t % D_;
    int l = (t / D_) % LTOT_;
    int b = t / ((size_t)D_ * LTOT_);
    int lv, st;
    if (l < 6400)      { lv = 0; st = 0; }
    else if (l < 8000) { lv = 1; st = 6400; }
    else if (l < 8400) { lv = 2; st = 8000; }
    else               { lv = 3; st = 8400; }
    int hw = c_lvlH[lv] * c_lvlW[lv];
    int sp = l - st;
    const float* sp_ptr = (lv == 0) ? s0 : (lv == 1) ? s1 : (lv == 2) ? s2 : s3;
    const float* pp_ptr = (lv == 0) ? p0 : (lv == 1) ? p1 : (lv == 2) ? p2 : p3;
    size_t idx = ((size_t)b * D_ + c) * hw + sp;
    x[t]   = sp_ptr[idx];
    pos[t] = pp_ptr[idx] + lvl_emb[lv * D_ + c];
}

// encoder reference points: ref[l] = ((x+0.5)/W, (y+0.5)/H)
__global__ void k_enc_ref(float* __restrict__ ref)
{
    int l = blockIdx.x * blockDim.x + threadIdx.x;
    if (l >= LTOT_) return;
    int lv;
    if (l < 6400) lv = 0; else if (l < 8000) lv = 1; else if (l < 8400) lv = 2; else lv = 3;
    int st = c_lvlS[lv], W = c_lvlW[lv], H = c_lvlH[lv];
    int sp = l - st;
    int y = sp / W, x = sp % W;
    ref[(size_t)l * 2 + 0] = (x + 0.5f) / W;
    ref[(size_t)l * 2 + 1] = (y + 0.5f) / H;
}

// refq = sigmoid(qp_w @ refp_w + refp_b), broadcast over batch
__global__ void k_refp(const float* __restrict__ qp_w, const float* __restrict__ w,
                       const float* __restrict__ bias, float* __restrict__ refq)
{
    int t = blockIdx.x * blockDim.x + threadIdx.x;
    if (t >= B_ * NQ_ * 2) return;
    int j = t & 1;
    int q = (t >> 1) % NQ_;
    float s = bias[j];
    for (int c = 0; c < D_; ++c) s += qp_w[(size_t)q * D_ + c] * w[c * 2 + j];
    refq[t] = 1.f / (1.f + __expf(-s));
}

// ---------------------------------------------------------------------------
// Deformable bilinear sampling. One 32-lane group per (b,q,h); lane = channel.
// ---------------------------------------------------------------------------
__global__ __launch_bounds__(256)
void k_deform_sample(const float* __restrict__ vbuf, const float* __restrict__ off,
                     const float* __restrict__ aw, const float* __restrict__ ref,
                     int refBstride, float* __restrict__ outp, int Qn, int Lv)
{
    int g    = blockIdx.x * 8 + (threadIdx.x >> 5);
    int lane = threadIdx.x & 31;
    int total = B_ * Qn * HEADS_;
    if (g >= total) return;
    int h = g % HEADS_;
    int q = (g / HEADS_) % Qn;
    int b = g / (HEADS_ * Qn);

    float rx = ref[(size_t)b * refBstride + (size_t)q * 2 + 0];
    float ry = ref[(size_t)b * refBstride + (size_t)q * 2 + 1];
    const float* offp = off + ((size_t)((size_t)b * Qn + q) * HEADS_ + h) * 32;
    const float* awp  = aw  + ((size_t)((size_t)b * Qn + q) * HEADS_ + h) * 16;

    float acc = 0.f;
    for (int l = 0; l < LVLS_; ++l) {
        int Hh = c_lvlH[l], Ww = c_lvlW[l], st = c_lvlS[l];
        const float* vb = vbuf + ((size_t)b * Lv + st) * D_ + h * HD_ + lane;
        for (int p = 0; p < PTS_; ++p) {
            float ox = offp[l * 8 + p * 2 + 0];
            float oy = offp[l * 8 + p * 2 + 1];
            float w  = awp[l * 4 + p];
            float x = (rx + ox / (float)Ww) * Ww - 0.5f;
            float y = (ry + oy / (float)Hh) * Hh - 0.5f;
            float x0 = floorf(x), y0 = floorf(y);
            float fx = x - x0, fy = y - y0;
            int x0i = (int)x0, y0i = (int)y0;
            for (int dy = 0; dy < 2; ++dy) {
                for (int dx = 0; dx < 2; ++dx) {
                    int xi = x0i + dx, yi = y0i + dy;
                    if (xi >= 0 && xi < Ww && yi >= 0 && yi < Hh) {
                        float cw = (dx ? fx : 1.f - fx) * (dy ? fy : 1.f - fy);
                        acc += w * cw * vb[(size_t)(yi * Ww + xi) * D_];
                    }
                }
            }
        }
    }
    outp[((size_t)((size_t)b * Qn + q)) * D_ + h * HD_ + lane] = acc;
}

// ---------------------------------------------------------------------------
// Decoder MHA attention core: one block per (b,h,q); 300 keys, HD=32.
// ---------------------------------------------------------------------------
__global__ __launch_bounds__(128)
void k_mha_attn(const float* __restrict__ qh, const float* __restrict__ kh,
                const float* __restrict__ vh, float* __restrict__ o)
{
    __shared__ float p[NQ_];
    __shared__ float red;
    int blk = blockIdx.x;
    int q = blk % NQ_;
    int h = (blk / NQ_) % HEADS_;
    int b = blk / (NQ_ * HEADS_);
    int tid = threadIdx.x;
    const float scale = 0.1767766952966369f; // 1/sqrt(32)

    const float* qrow = qh + ((size_t)b * NQ_ + q) * D_ + h * HD_;
    for (int k = tid; k < NQ_; k += 128) {
        const float* krow = kh + ((size_t)b * NQ_ + k) * D_ + h * HD_;
        float s = 0.f;
        for (int c = 0; c < HD_; ++c) s += qrow[c] * krow[c];
        p[k] = s * scale;
    }
    __syncthreads();
    if (tid == 0) {
        float mx = -1e30f;
        for (int k = 0; k < NQ_; ++k) mx = fmaxf(mx, p[k]);
        float sm = 0.f;
        for (int k = 0; k < NQ_; ++k) { float e = __expf(p[k] - mx); p[k] = e; sm += e; }
        red = 1.f / sm;
    }
    __syncthreads();
    if (tid < HD_) {
        float inv = red;
        float accv = 0.f;
        for (int k = 0; k < NQ_; ++k)
            accv += p[k] * vh[((size_t)b * NQ_ + k) * D_ + h * HD_ + tid];
        o[((size_t)b * NQ_ + q) * D_ + h * HD_ + tid] = accv * inv;
    }
}

// ---------------------------------------------------------------------------
// Host orchestration. Param leaf indices (JAX pytree: dict keys sorted, DFS):
//   0..7 src/pos interleaved, 8 query_embed_w, 9 query_pos_w,
//   dec[i] 22 leaves, enc[i] 16 leaves, level_embed, refp.b, refp.w
// ---------------------------------------------------------------------------
#define P_DEC(i, k) (10 + 22 * (i) + (k))
#define P_ENC(i, k) (142 + 16 * (i) + (k))
#define P_LVLEMB 238
#define P_REFP_B 239
#define P_REFP_W 240

struct GemmCtx {
    __bf16* A16;
    __bf16* W16;
    hipStream_t s;
};

static inline void cvtA(const GemmCtx& c, const float* A, size_t n)
{
    k_cvt_a<<<4096, 256, 0, c.s>>>(A, c.A16, n);
}

static inline void gemm16(const GemmCtx& c, const float* W, const float* bias,
                          float* Cm, int M, int N, int K, int wt, int act)
{
    k_cvt_w<<<512, 256, 0, c.s>>>(W, c.W16, K, N, wt);
    dim3 grid(N / BN, (M + BM - 1) / BM);
    gemm_wmma<<<grid, 256, 0, c.s>>>(c.A16, c.W16, bias, Cm, M, N, K, act);
}

extern "C" void kernel_launch(void* const* d_in, const int* in_sizes, int n_in,
                              void* d_out, int out_size, void* d_ws, size_t ws_size,
                              hipStream_t stream)
{
    (void)in_sizes; (void)n_in; (void)out_size; (void)ws_size;
    auto IN = [&](int i) { return (const float*)d_in[i]; };

    const size_t M_ENC = (size_t)B_ * LTOT_;       // 17000
    const size_t M_DEC = (size_t)B_ * NQ_;         // 600

    float* ws = (float*)d_ws;
    float* X    = ws;                               // M_ENC*256
    float* POS  = X    + M_ENC * D_;
    float* QB   = POS  + M_ENC * D_;                // q = x+pos; FFN hidden overlays here
    float* VB   = QB   + M_ENC * D_;
    float* OFFB = VB   + M_ENC * D_;
    float* AWB  = OFFB + M_ENC * D_;                // M_ENC*128
    float* SAMP = AWB  + M_ENC * 128;
    float* TMP  = SAMP + M_ENC * D_;
    float* HID  = QB;                               // M_ENC*1024 overlays dead bufs
    float* REF  = TMP  + M_ENC * D_;                // LTOT*2
    float* DOUT  = REF   + (size_t)LTOT_ * 2;
    float* DQPOS = DOUT  + M_DEC * D_;
    float* DQK   = DQPOS + M_DEC * D_;
    float* DQH   = DQK   + M_DEC * D_;
    float* DKH   = DQH   + M_DEC * D_;
    float* DVH   = DKH   + M_DEC * D_;
    float* DO_   = DVH   + M_DEC * D_;
    float* DTMP  = DO_   + M_DEC * D_;
    float* DOFF  = DTMP  + M_DEC * D_;
    float* DAW   = DOFF  + M_DEC * D_;
    float* DHID  = DAW   + M_DEC * 128;
    float* DREFQ = DHID  + M_DEC * DFFN_;
    __bf16* A16  = (__bf16*)(DREFQ + (size_t)B_ * NQ_ * 2);
    __bf16* W16  = A16 + M_ENC * DFFN_;             // max M*K elements

    GemmCtx G{A16, W16, stream};

    // ---- flatten inputs + encoder reference points ----
    {
        size_t total = M_ENC * D_;
        k_flatten<<<(total + 255) / 256, 256, 0, stream>>>(
            IN(0), IN(2), IN(4), IN(6), IN(1), IN(3), IN(5), IN(7),
            IN(P_LVLEMB), X, POS);
        k_enc_ref<<<(LTOT_ + 255) / 256, 256, 0, stream>>>(REF);
    }

    // ---- encoder ----
    for (int i = 0; i < 6; ++i) {
        const float* attw_b  = IN(P_ENC(i, 0)); const float* attw_w  = IN(P_ENC(i, 1));
        const float* off_b   = IN(P_ENC(i, 2)); const float* off_w   = IN(P_ENC(i, 3));
        const float* oproj_b = IN(P_ENC(i, 4)); const float* oproj_w = IN(P_ENC(i, 5));
        const float* vproj_b = IN(P_ENC(i, 6)); const float* vproj_w = IN(P_ENC(i, 7));
        const float* l1_b = IN(P_ENC(i, 8));  const float* l1_w = IN(P_ENC(i, 9));
        const float* l2_b = IN(P_ENC(i, 10)); const float* l2_w = IN(P_ENC(i, 11));
        const float* n1_b = IN(P_ENC(i, 12)); const float* n1_g = IN(P_ENC(i, 13));
        const float* n2_b = IN(P_ENC(i, 14)); const float* n2_g = IN(P_ENC(i, 15));

        k_add<<<2048, 256, 0, stream>>>(QB, X, POS, M_ENC * D_);
        cvtA(G, X, M_ENC * D_);
        gemm16(G, vproj_w, vproj_b, VB, (int)M_ENC, D_, D_, 0, 0);
        cvtA(G, QB, M_ENC * D_);
        gemm16(G, off_w,  off_b,  OFFB, (int)M_ENC, D_, D_, 0, 0);
        gemm16(G, attw_w, attw_b, AWB,  (int)M_ENC, 128, D_, 0, 0);
        k_softmax16<<<(int)((M_ENC * HEADS_ + 255) / 256), 256, 0, stream>>>(AWB, (int)(M_ENC * HEADS_));
        {
            int groups = (int)(B_ * LTOT_ * HEADS_);
            k_deform_sample<<<(groups + 7) / 8, 256, 0, stream>>>(
                VB, OFFB, AWB, REF, 0, SAMP, LTOT_, LTOT_);
        }
        cvtA(G, SAMP, M_ENC * D_);
        gemm16(G, oproj_w, oproj_b, TMP, (int)M_ENC, D_, D_, 0, 0);
        k_ln_residual<<<(int)((M_ENC + 7) / 8), 256, 0, stream>>>(X, TMP, n1_g, n1_b, (int)M_ENC);
        cvtA(G, X, M_ENC * D_);
        gemm16(G, l1_w, l1_b, HID, (int)M_ENC, DFFN_, D_, 0, 1);
        cvtA(G, HID, M_ENC * DFFN_);
        gemm16(G, l2_w, l2_b, TMP, (int)M_ENC, D_, DFFN_, 0, 0);
        k_ln_residual<<<(int)((M_ENC + 7) / 8), 256, 0, stream>>>(X, TMP, n2_g, n2_b, (int)M_ENC);
    }
    // memory = X

    // ---- decoder setup ----
    k_broadcast<<<600, 256, 0, stream>>>(DOUT,  IN(8), (size_t)NQ_ * D_, M_DEC * D_);
    k_broadcast<<<600, 256, 0, stream>>>(DQPOS, IN(9), (size_t)NQ_ * D_, M_DEC * D_);
    k_refp<<<(B_ * NQ_ * 2 + 255) / 256, 256, 0, stream>>>(IN(9), IN(P_REFP_W), IN(P_REFP_B), DREFQ);

    // ---- decoder ----
    for (int i = 0; i < 6; ++i) {
        const float* c_attw_b  = IN(P_DEC(i, 0)); const float* c_attw_w  = IN(P_DEC(i, 1));
        const float* c_off_b   = IN(P_DEC(i, 2)); const float* c_off_w   = IN(P_DEC(i, 3));
        const float* c_oproj_b = IN(P_DEC(i, 4)); const float* c_oproj_w = IN(P_DEC(i, 5));
        const float* c_vproj_b = IN(P_DEC(i, 6)); const float* c_vproj_w = IN(P_DEC(i, 7));
        const float* l1_b = IN(P_DEC(i, 8));  const float* l1_w = IN(P_DEC(i, 9));
        const float* l2_b = IN(P_DEC(i, 10)); const float* l2_w = IN(P_DEC(i, 11));
        const float* n1_b = IN(P_DEC(i, 12)); const float* n1_g = IN(P_DEC(i, 13));
        const float* n2_b = IN(P_DEC(i, 14)); const float* n2_g = IN(P_DEC(i, 15));
        const float* n3_b = IN(P_DEC(i, 16)); const float* n3_g = IN(P_DEC(i, 17));
        const float* inb  = IN(P_DEC(i, 18)); const float* inw  = IN(P_DEC(i, 19));
        const float* outb = IN(P_DEC(i, 20)); const float* outw = IN(P_DEC(i, 21));

        // self-attention: qk = out + qpos; qh,kh from qk; vh from out (W is [N,K])
        k_add<<<600, 256, 0, stream>>>(DQK, DOUT, DQPOS, M_DEC * D_);
        cvtA(G, DQK, M_DEC * D_);
        gemm16(G, inw,             inb,       DQH, (int)M_DEC, D_, D_, 1, 0);
        gemm16(G, inw + 256 * 256, inb + 256, DKH, (int)M_DEC, D_, D_, 1, 0);
        cvtA(G, DOUT, M_DEC * D_);
        gemm16(G, inw + 512 * 256, inb + 512, DVH, (int)M_DEC, D_, D_, 1, 0);
        k_mha_attn<<<B_ * HEADS_ * NQ_, 128, 0, stream>>>(DQH, DKH, DVH, DO_);
        cvtA(G, DO_, M_DEC * D_);
        gemm16(G, outw, outb, DTMP, (int)M_DEC, D_, D_, 1, 0);
        k_ln_residual<<<(int)((M_DEC + 7) / 8), 256, 0, stream>>>(DOUT, DTMP, n2_g, n2_b, (int)M_DEC);

        // cross deformable attention against encoder memory
        k_add<<<600, 256, 0, stream>>>(DQK, DOUT, DQPOS, M_DEC * D_);
        cvtA(G, X, M_ENC * D_);
        gemm16(G, c_vproj_w, c_vproj_b, VB, (int)M_ENC, D_, D_, 0, 0);
        cvtA(G, DQK, M_DEC * D_);
        gemm16(G, c_off_w,  c_off_b,  DOFF, (int)M_DEC, D_, D_, 0, 0);
        gemm16(G, c_attw_w, c_attw_b, DAW,  (int)M_DEC, 128, D_, 0, 0);
        k_softmax16<<<(int)((M_DEC * HEADS_ + 255) / 256), 256, 0, stream>>>(DAW, (int)(M_DEC * HEADS_));
        {
            int groups = (int)(B_ * NQ_ * HEADS_);
            k_deform_sample<<<(groups + 7) / 8, 256, 0, stream>>>(
                VB, DOFF, DAW, DREFQ, NQ_ * 2, DO_, NQ_, LTOT_);
        }
        cvtA(G, DO_, M_DEC * D_);
        gemm16(G, c_oproj_w, c_oproj_b, DTMP, (int)M_DEC, D_, D_, 0, 0);
        k_ln_residual<<<(int)((M_DEC + 7) / 8), 256, 0, stream>>>(DOUT, DTMP, n1_g, n1_b, (int)M_DEC);

        // FFN
        cvtA(G, DOUT, M_DEC * D_);
        gemm16(G, l1_w, l1_b, DHID, (int)M_DEC, DFFN_, D_, 0, 1);
        cvtA(G, DHID, M_DEC * DFFN_);
        gemm16(G, l2_w, l2_b, DTMP, (int)M_DEC, D_, DFFN_, 0, 0);
        k_ln_residual<<<(int)((M_DEC + 7) / 8), 256, 0, stream>>>(DOUT, DTMP, n3_g, n3_b, (int)M_DEC);
    }

    // ---- write outputs: (out, refq) concatenated ----
    float* outp = (float*)d_out;
    k_copy<<<600, 256, 0, stream>>>(outp, DOUT, M_DEC * D_);
    k_copy<<<8, 256, 0, stream>>>(outp + M_DEC * D_, DREFQ, (size_t)B_ * NQ_ * 2);
}
